// MultiHeadAttention_57080115364042
// MI455X (gfx1250) — matrix operand order repaired and checked by reading, not compiled, not run on hardware
//
#include <hip/hip_runtime.h>

// ---------------------------------------------------------------------------
// Transformer-XL attention for MI455X (gfx1250), wave32 + WMMA bf16.
//
// Math restructuring vs reference:
//   ac + ce = (q_h + u) . k_h        (fold u_bias into q)
//   bd + de = (q_h + v) . r_h        (fold v_bias into q)
//   br shifts scores uniformly along softmax axis -> no effect -> dropped.
//   bd+de computed WITHOUT materializing r_h:
//     P[b,h,i,:] = qv[b,h,i,:] @ Wr_h          (~4 GFLOP)
//     (bd+de)[b,h,i,j] = P[b,h,i,:] . r[i,j,:] (~34 GFLOP, streams r once)
//   This removes the 550 GFLOP r-projection and its 1 GiB intermediate.
//
// CDNA5 paths: v_wmma_f32_16x16x32_bf16 everywhere,
// global_load_async_to_lds_b128 (ASYNCcnt) for bf16 tiles, v_perm_b32 packing.
// ---------------------------------------------------------------------------

#define BB 4
#define SS 512
#define DD 1024
#define HH 16
#define DH 64

typedef __attribute__((ext_vector_type(16))) __bf16 bf16x16;
typedef __attribute__((ext_vector_type(2)))  __bf16 bf16x2;
typedef __attribute__((ext_vector_type(8)))  float  f32x8;
typedef __attribute__((__vector_size__(16))) int    v4i_t;

struct Bits256 { uint4 lo, hi; };

#if defined(__has_builtin)
#if __has_builtin(__builtin_amdgcn_cvt_pk_bf16_f32)
#define HAVE_PK_BF16 1
#endif
#if __has_builtin(__builtin_amdgcn_perm)
#define HAVE_PERM 1
#endif
#if __has_builtin(__builtin_amdgcn_global_load_async_to_lds_b128) && \
    __has_builtin(__builtin_amdgcn_s_wait_asynccnt)
#define USE_ASYNC_LDS 1
#endif
#endif
#ifndef HAVE_PK_BF16
#define HAVE_PK_BF16 0
#endif
#ifndef HAVE_PERM
#define HAVE_PERM 0
#endif
#ifndef USE_ASYNC_LDS
#define USE_ASYNC_LDS 0
#endif

__device__ __forceinline__ unsigned short f32_to_bf16(float f) {
  // round-half-up: 2 VALU ops; ties-vs-RNE delta is below bf16 quantization
  unsigned u = __builtin_bit_cast(unsigned, f);
  return (unsigned short)((u + 0x8000u) >> 16);
}
__device__ __forceinline__ unsigned pack2(float a, float b) {
#if HAVE_PK_BF16
  bf16x2 p = __builtin_amdgcn_cvt_pk_bf16_f32(a, b);
  return __builtin_bit_cast(unsigned, p);
#elif HAVE_PERM
  unsigned ua = __builtin_bit_cast(unsigned, a) + 0x8000u;
  unsigned ub = __builtin_bit_cast(unsigned, b) + 0x8000u;
  // dest bytes {0,1} = ua bytes {2,3}; dest bytes {2,3} = ub bytes {6,7 of s0}
  return __builtin_amdgcn_perm(ub, ua, 0x07060302u);
#else
  return (unsigned)f32_to_bf16(a) | ((unsigned)f32_to_bf16(b) << 16);
#endif
}

// stage 16 contiguous f32 -> 16 bf16 in LDS (two 16B stores)
__device__ __forceinline__ void cvt_store16(const float* __restrict__ src, unsigned short* dst) {
  const float4* s4 = reinterpret_cast<const float4*>(src);
  float4 f0 = s4[0], f1 = s4[1], f2 = s4[2], f3 = s4[3];
  uint4 p0, p1;
  p0.x = pack2(f0.x, f0.y); p0.y = pack2(f0.z, f0.w);
  p0.z = pack2(f1.x, f1.y); p0.w = pack2(f1.z, f1.w);
  p1.x = pack2(f2.x, f2.y); p1.y = pack2(f2.z, f2.w);
  p1.z = pack2(f3.x, f3.y); p1.w = pack2(f3.z, f3.w);
  *reinterpret_cast<uint4*>(dst)     = p0;
  *reinterpret_cast<uint4*>(dst + 8) = p1;
}

// stage 16 bf16 global -> LDS; async DMA path (ASYNCcnt) when available
__device__ __forceinline__ void stage16_bf16(const unsigned short* __restrict__ src,
                                             unsigned short* dst) {
#if USE_ASYNC_LDS
  typedef __attribute__((address_space(1))) v4i_t* gp_t;  // global (device) ptr
  typedef __attribute__((address_space(3))) v4i_t* lp_t;  // LDS ptr (32-bit)
  gp_t g0 = (gp_t)(unsigned long long)(const void*)src;
  gp_t g1 = (gp_t)(unsigned long long)(const void*)(src + 8);
  lp_t l0 = (lp_t)(unsigned)(unsigned long long)(void*)dst;
  lp_t l1 = (lp_t)(unsigned)(unsigned long long)(void*)(dst + 8);
  __builtin_amdgcn_global_load_async_to_lds_b128(g0, l0, 0, 0);
  __builtin_amdgcn_global_load_async_to_lds_b128(g1, l1, 0, 0);
#else
  const uint4* s = reinterpret_cast<const uint4*>(src);
  *reinterpret_cast<uint4*>(dst)     = s[0];
  *reinterpret_cast<uint4*>(dst + 8) = s[1];
#endif
}
__device__ __forceinline__ void stage_fence() {
#if USE_ASYNC_LDS
  __builtin_amdgcn_s_wait_asynccnt(0);
#endif
}

__device__ __forceinline__ bf16x16 frag_join(const unsigned short* p0, const unsigned short* p1) {
  Bits256 t;
  t.lo = *reinterpret_cast<const uint4*>(p0);
  t.hi = *reinterpret_cast<const uint4*>(p1);
  return __builtin_bit_cast(bf16x16, t);
}

// A fragment (16x32 bf16): lane<16 -> row=lane, K 0..15;
// lane>=16 -> row=lane-16, K 16..31
__device__ __forceinline__ bf16x16 load_a_frag(const unsigned short* As, int stride,
                                               int m0, int k0, int lane) {
  int half = lane >> 4;
  const unsigned short* base = As + (m0 + (lane & 15)) * stride + k0 + half * 16;
  return frag_join(base, base + 8);
}
// B fragment (32x16 bf16), stored in LDS as Bs[col][k] (K contiguous):
// lane<16 -> col=lane, K 0..15; lane>=16 -> col=lane-16, K 16..31
__device__ __forceinline__ bf16x16 load_b_frag(const unsigned short* Bs, int stride,
                                               int n0, int k0, int lane) {
  int half = lane >> 4;
  const unsigned short* base = Bs + (n0 + (lane & 15)) * stride + k0 + half * 16;
  return frag_join(base, base + 8);
}

__device__ __forceinline__ f32x8 wmma_bf16(bf16x16 a, bf16x16 b, f32x8 c) {
  return __builtin_amdgcn_wmma_f32_16x16x32_bf16(false, a, false, b, (short)0, c, false, false);
}

__device__ __forceinline__ float half_reduce_max(float x) {
  for (int s = 1; s < 16; s <<= 1) x = fmaxf(x, __shfl_xor(x, s, 32));
  return x;
}
__device__ __forceinline__ float half_reduce_sum(float x) {
  for (int s = 1; s < 16; s <<= 1) x += __shfl_xor(x, s, 32);
  return x;
}

// ---------------------------------------------------------------------------
// Generic  C[m,n] = sum_k A[m,k] * Bw[n,k]  (+ bias[n])  — i.e. X @ W^T + b
// Tile 128x128x32, 8 waves (4M x 2N), per-wave 32x64 (2x4 WMMA frags).
// ---------------------------------------------------------------------------
template <bool A_BF16, bool OUT_BF16>
__global__ __launch_bounds__(256) void gemm_xwT_kernel(
    const void* __restrict__ Ap, int lda, int a_co,
    const unsigned short* __restrict__ Bw, int ldb, int b_co,
    const float* __restrict__ bias,
    void* __restrict__ Outp, int ldo, int o_co,
    int K) {
  constexpr int BK = 32, SPAD = 40;  // 80B row pitch: conflict-free 16B frag reads
  __shared__ alignas(16) unsigned short As[128 * SPAD];
  __shared__ alignas(16) unsigned short Bs[128 * SPAD];

  const int tid = threadIdx.x, lane = tid & 31, wave = tid >> 5;
  const int wm = wave >> 1, wn = wave & 1;
  const int m0 = blockIdx.y * 128, n0 = blockIdx.x * 128;

  f32x8 acc[2][4] = {};

  const int srow = tid >> 1, skc = (tid & 1) * 16;
  for (int k0 = 0; k0 < K; k0 += BK) {
    // B tile: bf16 weights -> async DMA straight into LDS
    const unsigned short* bsrc = Bw + (size_t)(n0 + srow) * ldb + b_co + k0 + skc;
    stage16_bf16(bsrc, &Bs[srow * SPAD + skc]);
    if constexpr (A_BF16) {
      const unsigned short* asrc =
          (const unsigned short*)Ap + (size_t)(m0 + srow) * lda + a_co + k0 + skc;
      stage16_bf16(asrc, &As[srow * SPAD + skc]);
    } else {
      const float* asrc = (const float*)Ap + (size_t)(m0 + srow) * lda + a_co + k0 + skc;
      cvt_store16(asrc, &As[srow * SPAD + skc]);
    }
    if (k0 + BK < K) {  // gfx1250 global_prefetch_b8 for next K tile
      __builtin_prefetch((const char*)Bw + ((size_t)(n0 + srow) * ldb + b_co + k0 + BK) * 2, 0, 1);
    }
    stage_fence();
    __syncthreads();
    bf16x16 a0 = load_a_frag(As, SPAD, wm * 32, 0, lane);
    bf16x16 a1 = load_a_frag(As, SPAD, wm * 32 + 16, 0, lane);
#pragma unroll
    for (int j = 0; j < 4; ++j) {
      bf16x16 bf = load_b_frag(Bs, SPAD, wn * 64 + j * 16, 0, lane);
      acc[0][j] = wmma_bf16(a0, bf, acc[0][j]);
      acc[1][j] = wmma_bf16(a1, bf, acc[1][j]);
    }
    __syncthreads();
  }
#pragma unroll
  for (int mi = 0; mi < 2; ++mi)
#pragma unroll
    for (int ni = 0; ni < 4; ++ni) {
      int col = n0 + wn * 64 + ni * 16 + (lane & 15);
      float bv = bias ? bias[col] : 0.f;
      int rbase = m0 + wm * 32 + mi * 16 + 8 * (lane >> 4);
#pragma unroll
      for (int g = 0; g < 8; ++g) {
        float v = acc[mi][ni][g] + bv;
        size_t off = (size_t)(rbase + g) * ldo + o_co + col;
        if constexpr (OUT_BF16) ((unsigned short*)Outp)[off] = f32_to_bf16(v);
        else                    ((float*)Outp)[off] = v;
      }
    }
}

// ---------------------------------------------------------------------------
// r-term: for fixed i, scores_r[(b,h), j] = P[(b,h,i),:1024] . r[i,j,:1024]
// Tile: M=64 (all b,h), N=128 j, K=1024. grid = (4 n-tiles, 512 i).
// Streams r (1 GiB f32) exactly once, converting to bf16 on the fly.
// ---------------------------------------------------------------------------
__global__ __launch_bounds__(256) void rterm_kernel(
    const unsigned short* __restrict__ P,  // (B,S,H,D) bf16
    const float* __restrict__ r,           // (S,S,D) f32
    float* __restrict__ scores_r) {        // (B,H,S,S) f32
  constexpr int SPAD = 40;
  __shared__ alignas(16) unsigned short As[64 * SPAD];
  __shared__ alignas(16) unsigned short Bs[128 * SPAD];
  const int i = blockIdx.y, n0 = blockIdx.x * 128;
  const int tid = threadIdx.x, lane = tid & 31, wave = tid >> 5;
  const int wm = wave >> 2, wn = wave & 3;  // 2M x 4N waves, per-wave 32x32

  f32x8 acc[2][2] = {};
  for (int k0 = 0; k0 < 1024; k0 += 32) {
    if (tid < 128) {  // A: 64 rows x 32, bf16 -> async DMA to LDS
      int rr = tid >> 1, kc = (tid & 1) * 16;
      int b = rr >> 4, h = rr & 15;
      const unsigned short* asrc =
          P + ((size_t)((b * SS + i) * HH + h)) * DD + k0 + kc;
      stage16_bf16(asrc, &As[rr * SPAD + kc]);
    }
    {  // B: 128 j rows x 32, f32 -> bf16 (v_perm packing)
      int row = tid >> 1, kc = (tid & 1) * 16;
      const float* bsrc = r + ((size_t)i * SS + n0 + row) * DD + k0 + kc;
      cvt_store16(bsrc, &Bs[row * SPAD + kc]);
      if (k0 + 32 < 1024)
        __builtin_prefetch((const char*)(bsrc + 32), 0, 1);
    }
    stage_fence();
    __syncthreads();
    bf16x16 a0 = load_a_frag(As, SPAD, wm * 32, 0, lane);
    bf16x16 a1 = load_a_frag(As, SPAD, wm * 32 + 16, 0, lane);
#pragma unroll
    for (int j = 0; j < 2; ++j) {
      bf16x16 bf = load_b_frag(Bs, SPAD, wn * 32 + j * 16, 0, lane);
      acc[0][j] = wmma_bf16(a0, bf, acc[0][j]);
      acc[1][j] = wmma_bf16(a1, bf, acc[1][j]);
    }
    __syncthreads();
  }
#pragma unroll
  for (int mi = 0; mi < 2; ++mi)
#pragma unroll
    for (int ni = 0; ni < 2; ++ni) {
      int j = n0 + wn * 32 + ni * 16 + (lane & 15);
      int rb = wm * 32 + mi * 16 + 8 * (lane >> 4);
#pragma unroll
      for (int g = 0; g < 8; ++g) {
        int rr = rb + g;  // rr == b*16+h
        scores_r[((size_t)rr * SS + i) * SS + j] = acc[mi][ni][g];
      }
    }
}

// ---------------------------------------------------------------------------
// scores + softmax: per (b,h), 32 i-rows x all 512 j.
//   scores = ( qu.k  +  scores_r ) / 8 ;  attn = softmax_j(scores)
// 8 waves = 2M x 4N; per-wave 16 x 128 (8 frags). K = dh = 64.
// ---------------------------------------------------------------------------
__global__ __launch_bounds__(256) void attn_scores_kernel(
    const float* __restrict__ QU,   // (B,S,D) f32  (q_h + u folded)
    const float* __restrict__ Kf,   // (B,S,D) f32
    float* __restrict__ attn) {     // (B,H,S,S): in = r-term, out = attn weights
  constexpr int SPAD = 40;
  __shared__ alignas(16) unsigned short As[32 * SPAD];
  __shared__ alignas(16) unsigned short Bs[512 * SPAD];
  __shared__ float redmax[2][16][4];
  __shared__ float redsum[2][16][4];

  const int bh = blockIdx.y, b = bh >> 4, h = bh & 15;
  const int i0 = blockIdx.x * 32;
  const int tid = threadIdx.x, lane = tid & 31, wave = tid >> 5;
  const int wm = wave >> 2, wn = wave & 3;

  f32x8 acc[8] = {};
  for (int k0 = 0; k0 < DH; k0 += 32) {
    if (tid < 64) {  // A: 32 rows x 32
      int rr = tid >> 1, kc = (tid & 1) * 16;
      const float* asrc = QU + ((size_t)(b * SS + i0 + rr)) * DD + h * DH + k0 + kc;
      cvt_store16(asrc, &As[rr * SPAD + kc]);
    }
#pragma unroll
    for (int p = 0; p < 4; ++p) {  // B: 512 rows x 32
      int idx = p * 256 + tid;
      int row = idx >> 1, kc = (idx & 1) * 16;
      const float* bsrc = Kf + ((size_t)(b * SS + row)) * DD + h * DH + k0 + kc;
      cvt_store16(bsrc, &Bs[row * SPAD + kc]);
    }
    __syncthreads();
    bf16x16 a = load_a_frag(As, SPAD, wm * 16, 0, lane);
#pragma unroll
    for (int f = 0; f < 8; ++f) {
      bf16x16 bf = load_b_frag(Bs, SPAD, wn * 128 + f * 16, 0, lane);
      acc[f] = wmma_bf16(a, bf, acc[f]);
    }
    __syncthreads();
  }

  // add r-term and scale by 1/sqrt(dh)
  const int rbase = i0 + wm * 16 + 8 * (lane >> 4);
  float x[8][8];
#pragma unroll
  for (int f = 0; f < 8; ++f) {
    int j = wn * 128 + f * 16 + (lane & 15);
#pragma unroll
    for (int g = 0; g < 8; ++g)
      x[f][g] = (acc[f][g] +
                 attn[((size_t)bh * SS + rbase + g) * SS + j]) * 0.125f;
  }
  // row max (per-lane over frags, then half-wave, then cross-wave via LDS)
  float rmax[8];
#pragma unroll
  for (int g = 0; g < 8; ++g) {
    float m = x[0][g];
#pragma unroll
    for (int f = 1; f < 8; ++f) m = fmaxf(m, x[f][g]);
    rmax[g] = half_reduce_max(m);
  }
  if ((lane & 15) == 0) {
#pragma unroll
    for (int g = 0; g < 8; ++g) redmax[wm][(lane >> 4) * 8 + g][wn] = rmax[g];
  }
  __syncthreads();
  float gmax[8], rsum[8];
#pragma unroll
  for (int g = 0; g < 8; ++g) {
    int row = (lane >> 4) * 8 + g;
    float m = redmax[wm][row][0];
    m = fmaxf(m, redmax[wm][row][1]);
    m = fmaxf(m, redmax[wm][row][2]);
    m = fmaxf(m, redmax[wm][row][3]);
    gmax[g] = m;
    float s = 0.f;
#pragma unroll
    for (int f = 0; f < 8; ++f) { x[f][g] = __expf(x[f][g] - m); s += x[f][g]; }
    rsum[g] = half_reduce_sum(s);
  }
  if ((lane & 15) == 0) {
#pragma unroll
    for (int g = 0; g < 8; ++g) redsum[wm][(lane >> 4) * 8 + g][wn] = rsum[g];
  }
  __syncthreads();
#pragma unroll
  for (int g = 0; g < 8; ++g) {
    int row = (lane >> 4) * 8 + g;
    float inv = 1.f / (redsum[wm][row][0] + redsum[wm][row][1] +
                       redsum[wm][row][2] + redsum[wm][row][3]);
#pragma unroll
    for (int f = 0; f < 8; ++f) {
      int j = wn * 128 + f * 16 + (lane & 15);
      attn[((size_t)bh * SS + rbase + g) * SS + j] = x[f][g] * inv;
    }
  }
}

// ---------------------------------------------------------------------------
// out_h = attn @ v_h : per (b,h), (512x512) @ (512x64).  Tile M=128,N=64,K=32.
// 8 waves = 4M x 2N; per-wave 32x32.
// ---------------------------------------------------------------------------
__global__ __launch_bounds__(256) void attn_v_kernel(
    const float* __restrict__ attn,        // (B,H,S,S) f32
    const unsigned short* __restrict__ VT, // (B,H,DH,S) bf16
    unsigned short* __restrict__ OH) {     // (B,S,D) bf16
  constexpr int SPAD = 40;
  __shared__ alignas(16) unsigned short As[128 * SPAD];
  __shared__ alignas(16) unsigned short Bs[64 * SPAD];
  const int bh = blockIdx.y, b = bh >> 4, h = bh & 15;
  const int i0 = blockIdx.x * 128;
  const int tid = threadIdx.x, lane = tid & 31, wave = tid >> 5;
  const int wm = wave >> 1, wn = wave & 1;

  f32x8 acc[2][2] = {};
  for (int k0 = 0; k0 < SS; k0 += 32) {
    {  // A: 128 rows x 32 (f32 attn -> bf16)
      int row = tid >> 1, kc = (tid & 1) * 16;
      const float* asrc = attn + ((size_t)bh * SS + i0 + row) * SS + k0 + kc;
      cvt_store16(asrc, &As[row * SPAD + kc]);
    }
    if (tid < 128) {  // B: 64 d-rows x 32, bf16 -> async DMA to LDS
      int row = tid >> 1, kc = (tid & 1) * 16;
      const unsigned short* bsrc = VT + ((size_t)bh * DH + row) * SS + k0 + kc;
      stage16_bf16(bsrc, &Bs[row * SPAD + kc]);
    }
    stage_fence();
    __syncthreads();
    bf16x16 a0 = load_a_frag(As, SPAD, wm * 32, 0, lane);
    bf16x16 a1 = load_a_frag(As, SPAD, wm * 32 + 16, 0, lane);
#pragma unroll
    for (int j = 0; j < 2; ++j) {
      bf16x16 bf = load_b_frag(Bs, SPAD, wn * 32 + j * 16, 0, lane);
      acc[0][j] = wmma_bf16(a0, bf, acc[0][j]);
      acc[1][j] = wmma_bf16(a1, bf, acc[1][j]);
    }
    __syncthreads();
  }
#pragma unroll
  for (int mi = 0; mi < 2; ++mi)
#pragma unroll
    for (int ni = 0; ni < 2; ++ni) {
      int d = wn * 32 + ni * 16 + (lane & 15);
      int rb = i0 + wm * 32 + mi * 16 + 8 * (lane >> 4);
#pragma unroll
      for (int g = 0; g < 8; ++g)
        OH[((size_t)(b * SS + rb + g)) * DD + h * DH + d] =
            f32_to_bf16(acc[mi][ni][g]);
    }
}

// ----------------------------- small elementwise ---------------------------
__global__ void cvt_bf16_kernel(const float* __restrict__ src,
                                unsigned short* __restrict__ dst, int n) {
  int idx = blockIdx.x * 256 + threadIdx.x;
  if (idx < n) dst[idx] = f32_to_bf16(src[idx]);
}
__global__ void wrT_kernel(const float* __restrict__ Wr,
                           unsigned short* __restrict__ WrT) {
  int idx = blockIdx.x * 256 + threadIdx.x;           // over 1M
  int n = idx >> 10, m = idx & 1023;
  WrT[idx] = f32_to_bf16(Wr[(size_t)m * DD + n]);
}
__global__ void addbias_kernel(const float* __restrict__ src,
                               const float* __restrict__ bias,
                               float* __restrict__ dst) {
  int idx = blockIdx.x * 256 + threadIdx.x;           // over B*S*D
  dst[idx] = src[idx] + bias[idx & (DD - 1)];
}
__global__ void vtrans_kernel(const float* __restrict__ Vf,
                              unsigned short* __restrict__ VT) {
  int idx = blockIdx.x * 256 + threadIdx.x;           // over B*S*D
  int j = idx & 511, d = (idx >> 9) & 63, bh = idx >> 15;
  int b = bh >> 4, h = bh & 15;
  VT[idx] = f32_to_bf16(Vf[((size_t)(b * SS + j)) * DD + h * DH + d]);
}

// ---------------------------------------------------------------------------
extern "C" void kernel_launch(void* const* d_in, const int* in_sizes, int n_in,
                              void* d_out, int out_size, void* d_ws, size_t ws_size,
                              hipStream_t stream) {
  (void)in_sizes; (void)n_in; (void)out_size; (void)ws_size;
  const float* q  = (const float*)d_in[0];
  const float* k  = (const float*)d_in[1];
  const float* v  = (const float*)d_in[2];
  const float* r  = (const float*)d_in[3];
  const float* Wq = (const float*)d_in[4];
  const float* bq = (const float*)d_in[5];
  const float* Wk = (const float*)d_in[6];
  const float* bk = (const float*)d_in[7];
  const float* Wv = (const float*)d_in[8];
  const float* bv = (const float*)d_in[9];
  const float* Wr = (const float*)d_in[10];
  /* br = d_in[11]: uniform shift along softmax axis -> no effect on outputs */
  const float* Wo = (const float*)d_in[12];
  const float* bo = (const float*)d_in[13];
  const float* ub = (const float*)d_in[14];
  const float* vb = (const float*)d_in[15];

  float* outp = (float*)d_out;                         // (B,S,D) attn_output
  float* attn = outp + (size_t)BB * SS * DD;           // (B,H,S,S) attn_weights

  // ---- workspace carve ----
  char* wp = (char*)d_ws;
  auto take = [&](size_t bytes) {
    void* p = wp; wp += (bytes + 255) & ~(size_t)255; return p;
  };
  const size_t NBSD = (size_t)BB * SS * DD;            // 2,097,152
  float* Qf = (float*)take(NBSD * 4);
  float* Kf = (float*)take(NBSD * 4);
  float* Vf = (float*)take(NBSD * 4);
  float* QU = (float*)take(NBSD * 4);
  float* QV = (float*)take(NBSD * 4);
  unsigned short* Wq_bf = (unsigned short*)take((size_t)DD * DD * 2);
  unsigned short* Wk_bf = (unsigned short*)take((size_t)DD * DD * 2);
  unsigned short* Wv_bf = (unsigned short*)take((size_t)DD * DD * 2);
  unsigned short* Wo_bf = (unsigned short*)take((size_t)DD * DD * 2);
  unsigned short* WrT_bf = (unsigned short*)take((size_t)DD * DD * 2);
  unsigned short* VT = (unsigned short*)take(NBSD * 2);
  unsigned short* OH = (unsigned short*)take(NBSD * 2);
  unsigned short* P  = (unsigned short*)take(NBSD * HH * 2);  // (B,S,H,D) 67MB

  const int WGRID = (DD * DD) / 256;                   // 4096
  cvt_bf16_kernel<<<WGRID, 256, 0, stream>>>(Wq, Wq_bf, DD * DD);
  cvt_bf16_kernel<<<WGRID, 256, 0, stream>>>(Wk, Wk_bf, DD * DD);
  cvt_bf16_kernel<<<WGRID, 256, 0, stream>>>(Wv, Wv_bf, DD * DD);
  cvt_bf16_kernel<<<WGRID, 256, 0, stream>>>(Wo, Wo_bf, DD * DD);
  wrT_kernel<<<WGRID, 256, 0, stream>>>(Wr, WrT_bf);

  // projections: (2048 x 1024) = X @ W^T + b, f32 out
  dim3 g_proj(DD / 128, (BB * SS) / 128);              // (8,16)
  gemm_xwT_kernel<false, false><<<g_proj, 256, 0, stream>>>(
      q, DD, 0, Wq_bf, DD, 0, bq, Qf, DD, 0, DD);
  gemm_xwT_kernel<false, false><<<g_proj, 256, 0, stream>>>(
      k, DD, 0, Wk_bf, DD, 0, bk, Kf, DD, 0, DD);
  gemm_xwT_kernel<false, false><<<g_proj, 256, 0, stream>>>(
      v, DD, 0, Wv_bf, DD, 0, bv, Vf, DD, 0, DD);

  const int EGRID = (int)(NBSD / 256);                 // 8192
  addbias_kernel<<<EGRID, 256, 0, stream>>>(Qf, ub, QU);
  addbias_kernel<<<EGRID, 256, 0, stream>>>(Qf, vb, QV);
  vtrans_kernel<<<EGRID, 256, 0, stream>>>(Vf, VT);

  // P[b,i,h,:] = qv[b,h,i,:] @ Wr_h   (per-head K=64 GEMM, bf16 out)
  for (int h = 0; h < HH; ++h) {
    gemm_xwT_kernel<false, true><<<g_proj, 256, 0, stream>>>(
        QV, DD, h * DH, WrT_bf, DD, h * DH, nullptr,
        P, HH * DD, h * DD, DH);
  }

  // r-term into the attn buffer (streams r exactly once)
  rterm_kernel<<<dim3(SS / 128, SS), 256, 0, stream>>>(P, r, attn);

  // scores + softmax (reads r-term from attn, writes attn weights in place)
  attn_scores_kernel<<<dim3(SS / 32, BB * HH), 256, 0, stream>>>(QU, Kf, attn);

  // out_h = attn @ v_h
  attn_v_kernel<<<dim3(SS / 128, BB * HH), 256, 0, stream>>>(attn, VT, OH);

  // final projection: attn_output = out_h @ Wo^T + bo  (f32 out to d_out)
  gemm_xwT_kernel<true, false><<<g_proj, 256, 0, stream>>>(
      OH, DD, 0, Wo_bf, DD, 0, bo, outp, DD, 0, DD);
}
